// SPAIRGlimpseRGBDecoder_15470472200212
// MI455X (gfx1250) — compile-verified, matrix operand order checked by reading, not run on hardware
//
#include <hip/hip_runtime.h>

typedef __attribute__((ext_vector_type(16))) _Float16 v16h;
typedef __attribute__((ext_vector_type(8)))  _Float16 v8h;
typedef __attribute__((ext_vector_type(8)))  float    v8f;
typedef __attribute__((ext_vector_type(4)))  float    v4f;

// ---------- helpers ----------

// Order-preserving encoding: unsigned compare order == float compare order.
__device__ __forceinline__ unsigned enc_ord(float f) {
    unsigned b = __float_as_uint(f);
    return (b & 0x80000000u) ? ~b : (b | 0x80000000u);
}
__device__ __forceinline__ float dec_ord(unsigned u) {
    unsigned b = (u & 0x80000000u) ? (u & 0x7fffffffu) : ~u;
    return __uint_as_float(b);
}
__device__ __forceinline__ float celuf(float x) {
    return x > 0.f ? x : (__expf(x) - 1.f);
}
// 16-bit A/B fragment K-index map (CDNA5 16x16x32 f16 layout):
// lanes 0-15 (g=0) hold K {0..7, 16..23}; lanes 16-31 (g=1) hold K {8..15, 24..31}
__device__ __forceinline__ int kmap(int e, int g) {
    return e + 8 * g + ((e >= 8) ? 8 : 0);
}

#define ENC_NEG_INF 0x007FFFFFu  // enc_ord(-inf)

// ---------- prep: swizzle f32 weights [Krows x N] into f16 B-fragment order ----------
// out[((kc*NT + nt)*32 + lane)*16 + e] = W[(kc*32 + kmap(e, lane>>4)) * N + nt*16 + (lane&15)]
__global__ void swizzle_w(const float* __restrict__ W, int Krows, int N, int KC,
                          _Float16* __restrict__ out) {
    int NT = N / 16;
    int total = KC * NT * 32 * 16;
    for (int i = blockIdx.x * blockDim.x + threadIdx.x; i < total;
         i += gridDim.x * blockDim.x) {
        int e    = i & 15;
        int lane = (i >> 4) & 31;
        int frag = i >> 9;
        int kc = frag / NT, nt = frag % NT;
        int g = lane >> 4, n = lane & 15;
        int k = kc * 32 + kmap(e, g);
        float v = (k < Krows) ? W[k * N + nt * 16 + n] : 0.f;
        out[i] = (_Float16)v;
    }
}

__global__ void fill_u32(unsigned* __restrict__ p, unsigned val, int n) {
    for (int i = blockIdx.x * blockDim.x + threadIdx.x; i < n;
         i += gridDim.x * blockDim.x)
        p[i] = val;
}

// ---------- main per-edge MLP + segment-max scatter ----------
// One wave = one tile of 16 edges. GEMM1: [16 x (C_IN+3)] @ W1 -> CELU ->
// LDS round-trip -> GEMM2: [16 x C_MID] @ W2 -> atomic-max scatter by dst.
template <int C_IN, int K1C, int C_MID, int K2C, int C_OUT, bool XF16>
__global__ void __launch_bounds__(128)
pointconv_edges(const float* __restrict__ xf, const _Float16* __restrict__ xh,
                const float* __restrict__ pos, const int* __restrict__ edge, int E_,
                const _Float16* __restrict__ B1, const float* __restrict__ b1,
                const _Float16* __restrict__ B2, const float* __restrict__ b2,
                unsigned* __restrict__ agg) {
    constexpr int NT1 = C_MID / 16;
    constexpr int NT2 = C_OUT / 16;
    constexpr int LD2 = K2C * 32;  // padded K for GEMM2

    __shared__ __align__(32) _Float16 hbuf[4][16 * LD2];

    const int lane = threadIdx.x & 31;
    const int wave = threadIdx.x >> 5;
    const int g = lane >> 4;
    const int mrow = lane & 15;
    const int n = lane & 15;

    const int ntiles = (E_ + 15) / 16;
    const int tile = blockIdx.x * 4 + wave;
    if (tile >= ntiles) return;  // uniform per wave -> EXEC stays all-ones for WMMA
    const int base = tile * 16;

    int eidx = base + mrow;
    if (eidx >= E_) eidx = E_ - 1;
    const int src_m = edge[eidx];        // src of the row this lane owns
    const int dst_m = edge[E_ + eidx];   // dst of the row this lane owns

    // ---- GEMM1: acc1 = [A | rel] @ W1 + b1 ----
    v8f acc1[NT1];
#pragma unroll
    for (int nt = 0; nt < NT1; ++nt) {
        float b = b1[nt * 16 + n];
#pragma unroll
        for (int v = 0; v < 8; ++v) acc1[nt][v] = b;
    }

#pragma unroll
    for (int kc = 0; kc < K1C; ++kc) {
        union { v16h v; _Float16 h[16]; } A;
        if ((kc + 1) * 32 <= C_IN) {
            // fast path: two contiguous 8-element runs per lane
            if (XF16) {
                v8h lo = *(const v8h*)(xh + (size_t)src_m * C_IN + kc * 32 + 8 * g);
                v8h hi = *(const v8h*)(xh + (size_t)src_m * C_IN + kc * 32 + 16 + 8 * g);
#pragma unroll
                for (int e = 0; e < 8; ++e) { A.h[e] = lo[e]; A.h[8 + e] = hi[e]; }
            } else {
                const float* p = xf + (size_t)src_m * C_IN + kc * 32 + 8 * g;
                v4f a0 = *(const v4f*)(p);      v4f a1 = *(const v4f*)(p + 4);
                v4f a2 = *(const v4f*)(p + 16); v4f a3 = *(const v4f*)(p + 20);
#pragma unroll
                for (int e = 0; e < 4; ++e) {
                    A.h[e]      = (_Float16)a0[e];
                    A.h[4 + e]  = (_Float16)a1[e];
                    A.h[8 + e]  = (_Float16)a2[e];
                    A.h[12 + e] = (_Float16)a3[e];
                }
            }
        } else {
            // boundary chunk: x tail + 3 rel-pos cols + zero pad
#pragma unroll
            for (int e = 0; e < 16; ++e) {
                int k = kc * 32 + kmap(e, g);
                float val;
                if (k < C_IN)
                    val = XF16 ? (float)xh[(size_t)src_m * C_IN + k]
                               : xf[(size_t)src_m * C_IN + k];
                else if (k < C_IN + 3) {
                    int d = k - C_IN;
                    val = pos[dst_m * 3 + d] - pos[src_m * 3 + d];
                } else
                    val = 0.f;
                A.h[e] = (_Float16)val;
            }
        }
#pragma unroll
        for (int nt = 0; nt < NT1; ++nt) {
            v16h B = *(const v16h*)(B1 + (((size_t)kc * NT1 + nt) * 32 + lane) * 16);
            acc1[nt] = __builtin_amdgcn_wmma_f32_16x16x32_f16(
                false, A.v, false, B, (short)0, acc1[nt], false, false);
        }
    }

    // ---- CELU + stage h into wave-private LDS (f16, K padded to LD2) ----
    _Float16* hw = hbuf[wave];
    if (C_MID < LD2) {
        for (int i = lane; i < 16 * LD2; i += 32) hw[i] = (_Float16)0.f;
    }
#pragma unroll
    for (int nt = 0; nt < NT1; ++nt) {
#pragma unroll
        for (int v = 0; v < 8; ++v) {
            int m = v + 8 * g;
            hw[m * LD2 + nt * 16 + n] = (_Float16)celuf(acc1[nt][v]);
        }
    }

    // ---- GEMM2: acc2 = h @ W2 + b2 ----
    v8f acc2[NT2];
#pragma unroll
    for (int nt = 0; nt < NT2; ++nt) {
        float b = b2[nt * 16 + n];
#pragma unroll
        for (int v = 0; v < 8; ++v) acc2[nt][v] = b;
    }
#pragma unroll
    for (int kc = 0; kc < K2C; ++kc) {
        union { v16h v; _Float16 h[16]; } A2;
        v8h lo = *(const v8h*)(hw + mrow * LD2 + kc * 32 + 8 * g);
        v8h hi = *(const v8h*)(hw + mrow * LD2 + kc * 32 + 16 + 8 * g);
#pragma unroll
        for (int e = 0; e < 8; ++e) { A2.h[e] = lo[e]; A2.h[8 + e] = hi[e]; }
#pragma unroll
        for (int nt = 0; nt < NT2; ++nt) {
            v16h B = *(const v16h*)(B2 + (((size_t)kc * NT2 + nt) * 32 + lane) * 16);
            acc2[nt] = __builtin_amdgcn_wmma_f32_16x16x32_f16(
                false, A2.v, false, B, (short)0, acc2[nt], false, false);
        }
    }

    // ---- segment_max scatter (ordered-uint atomic max) ----
#pragma unroll
    for (int v = 0; v < 8; ++v) {
        int m = v + 8 * g;
        if (base + m < E_) {
            int row = edge[E_ + base + m];
#pragma unroll
            for (int nt = 0; nt < NT2; ++nt) {
                atomicMax(&agg[(size_t)row * C_OUT + nt * 16 + n],
                          enc_ord(acc2[nt][v]));
            }
        }
    }
}

// ---------- finalize: decode, empty-segment -> 0, CELU, store f16 features ----------
__global__ void finalize_to_f16(const unsigned* __restrict__ agg,
                                _Float16* __restrict__ xout, int total) {
    for (int i = blockIdx.x * blockDim.x + threadIdx.x; i < total;
         i += gridDim.x * blockDim.x) {
        float f = dec_ord(agg[i]);
        if (!__builtin_isfinite(f)) f = 0.f;
        xout[i] = (_Float16)celuf(f);
    }
}

// ---------- last stage: celu(agg3) @ lin_W + lin_b ----------
__global__ void finalize_last(const unsigned* __restrict__ agg,
                              const float* __restrict__ linW,
                              const float* __restrict__ linb,
                              float* __restrict__ out, int Nn) {
    int i = blockIdx.x * blockDim.x + threadIdx.x;
    if (i >= Nn * 3) return;
    int nd = i / 3, j = i % 3;
    float s = linb[j];
#pragma unroll
    for (int c = 0; c < 16; ++c) {
        float f = dec_ord(agg[(size_t)nd * 16 + c]);
        if (!__builtin_isfinite(f)) f = 0.f;
        s += celuf(f) * linW[c * 3 + j];
    }
    out[i] = s;
}

// ---------- launcher ----------
extern "C" void kernel_launch(void* const* d_in, const int* in_sizes, int n_in,
                              void* d_out, int out_size, void* d_ws, size_t ws_size,
                              hipStream_t stream) {
    const float* z_what = (const float*)d_in[0];
    const float* pos1 = (const float*)d_in[1];
    const float* pos2 = (const float*)d_in[2];
    const float* pos3 = (const float*)d_in[3];
    const int* edge1 = (const int*)d_in[4];
    const int* edge2 = (const int*)d_in[5];
    const int* edge3 = (const int*)d_in[6];
    const float* c1W1 = (const float*)d_in[7];  const float* c1b1 = (const float*)d_in[8];
    const float* c1W2 = (const float*)d_in[9];  const float* c1b2 = (const float*)d_in[10];
    const float* c2W1 = (const float*)d_in[11]; const float* c2b1 = (const float*)d_in[12];
    const float* c2W2 = (const float*)d_in[13]; const float* c2b2 = (const float*)d_in[14];
    const float* c3W1 = (const float*)d_in[15]; const float* c3b1 = (const float*)d_in[16];
    const float* c3W2 = (const float*)d_in[17]; const float* c3b2 = (const float*)d_in[18];
    const float* linW = (const float*)d_in[19]; const float* linb = (const float*)d_in[20];

    const int N = in_sizes[0] / 128;
    const int E = in_sizes[4] / 2;

    // workspace layout
    char* ws = (char*)d_ws;
    size_t off = 0;
    auto take = [&](size_t bytes) {
        size_t o = off;
        off = (off + bytes + 255) & ~(size_t)255;
        return o;
    };
    _Float16* B11 = (_Float16*)(ws + take(20480 * 2));  // 5 chunks x 8 ntiles
    _Float16* B12 = (_Float16*)(ws + take(8192 * 2));   // 4 x 4
    _Float16* B21 = (_Float16*)(ws + take(3072 * 2));   // 3 x 2
    _Float16* B22 = (_Float16*)(ws + take(1024 * 2));   // 1 x 2
    _Float16* B31 = (_Float16*)(ws + take(1024 * 2));   // 2 x 1
    _Float16* B32 = (_Float16*)(ws + take(512 * 2));    // 1 x 1
    unsigned* agg1 = (unsigned*)(ws + take((size_t)N * 112 * 4));  // 64+32+16 contiguous
    unsigned* agg2 = agg1 + (size_t)N * 64;
    unsigned* agg3 = agg2 + (size_t)N * 32;
    _Float16* x2 = (_Float16*)(ws + take((size_t)N * 64 * 2));
    _Float16* x3 = (_Float16*)(ws + take((size_t)N * 32 * 2));

    // init segment-max accumulators to enc(-inf)
    fill_u32<<<1024, 256, 0, stream>>>(agg1, ENC_NEG_INF, N * 112);

    // pre-swizzle weights to f16 fragment layout
    swizzle_w<<<80, 256, 0, stream>>>(c1W1, 131, 128, 5, B11);
    swizzle_w<<<32, 256, 0, stream>>>(c1W2, 128, 64, 4, B12);
    swizzle_w<<<12, 256, 0, stream>>>(c2W1, 67, 32, 3, B21);
    swizzle_w<<<4, 256, 0, stream>>>(c2W2, 32, 32, 1, B22);
    swizzle_w<<<4, 256, 0, stream>>>(c3W1, 35, 16, 2, B31);
    swizzle_w<<<2, 256, 0, stream>>>(c3W2, 16, 16, 1, B32);

    const int ntiles = (E + 15) / 16;
    const int blocks = (ntiles + 3) / 4;  // 4 waves/block, 1 tile/wave

    pointconv_edges<128, 5, 128, 4, 64, false><<<blocks, 128, 0, stream>>>(
        z_what, nullptr, pos1, edge1, E, B11, c1b1, B12, c1b2, agg1);
    finalize_to_f16<<<2048, 256, 0, stream>>>(agg1, x2, N * 64);

    pointconv_edges<64, 3, 32, 1, 32, true><<<blocks, 128, 0, stream>>>(
        nullptr, x2, pos2, edge2, E, B21, c2b1, B22, c2b2, agg2);
    finalize_to_f16<<<1024, 256, 0, stream>>>(agg2, x3, N * 32);

    pointconv_edges<32, 2, 16, 1, 16, true><<<blocks, 128, 0, stream>>>(
        nullptr, x3, pos3, edge3, E, B31, c3b1, B32, c3b2, agg3);

    finalize_last<<<(N * 3 + 255) / 256, 256, 0, stream>>>(agg3, linW, linb,
                                                           (float*)d_out, N);
}
